// MultiHeadLatentAttention_38087769981544
// MI455X (gfx1250) — compile-verified, matrix operand order checked by reading, not run on hardware
//
#include <hip/hip_runtime.h>
#include <hip/hip_bf16.h>
#include <stdint.h>
#include <stddef.h>

// ---------------------------------------------------------------------------
// CDNA5 / gfx1250 MLA forward.  bf16 WMMA (v_wmma_f32_16x16x32_bf16), f32 acc.
// wave32; 16x16x32 fragments; A and B^T share one K-contiguous lane layout.
// GEMMs use register-staged double-buffered LDS tiles; fragment loads are
// batched so WMMAs issue back-to-back behind a single s_wait_dscnt.
// ---------------------------------------------------------------------------

typedef __attribute__((ext_vector_type(16))) __bf16 v16bf;
typedef __attribute__((ext_vector_type(8)))  float  v8f;
typedef unsigned int u32x4 __attribute__((ext_vector_type(4)));

union Frag {
  v16bf v;
  u32x4 q[2];
};

__device__ __forceinline__ unsigned short f2bf(float f) {
  union { float f; unsigned u; } x; x.f = f;
  unsigned r = x.u + 0x7FFFu + ((x.u >> 16) & 1u);   // round-to-nearest-even
  return (unsigned short)(r >> 16);
}
__device__ __forceinline__ float bf2f(unsigned short h) {
  union { unsigned u; float f; } x; x.u = ((unsigned)h) << 16; return x.f;
}

// Load one 16x32 bf16 fragment in the CDNA5 A-operand layout (also used for
// B given N-major [N][K] storage).  lane g=lane>>4 picks the K half-groups:
// k in [8g, 8g+8) -> VGPRs 0..3, k in [16+8g, 16+8g+8) -> VGPRs 4..7.
// Two ds_load_b128 per lane.
__device__ __forceinline__ void load_frag(Frag& f, const unsigned short* base,
                                          int stride, int row0, int k0, int lane) {
  const int g = lane >> 4, m = lane & 15;
  const unsigned short* p = base + (size_t)(row0 + m) * stride + k0 + 8 * g;
  f.q[0] = *(const u32x4*)(const void*)p;
  f.q[1] = *(const u32x4*)(const void*)(p + 16);
}

__device__ __forceinline__ v8f wmma_bf16(const Frag& a, const Frag& b, v8f c) {
  return __builtin_amdgcn_wmma_f32_16x16x32_bf16(false, a.v, false, b.v,
                                                 (short)0, c, false, false);
}

// ---------------------------------------------------------------------------
// elementwise f32 -> bf16 cast
// ---------------------------------------------------------------------------
__global__ __launch_bounds__(256) void cast_f32_bf16_kernel(
    const float* __restrict__ in, unsigned short* __restrict__ out, int n) {
  int i = (blockIdx.x * 256 + threadIdx.x) * 4;
  if (i + 3 < n) {
    out[i + 0] = f2bf(in[i + 0]);
    out[i + 1] = f2bf(in[i + 1]);
    out[i + 2] = f2bf(in[i + 2]);
    out[i + 3] = f2bf(in[i + 3]);
  }
}

// ---------------------------------------------------------------------------
// f32 [K][N] -> bf16 [N][K]  (weights -> B^T layout for WMMA B fragments)
// ---------------------------------------------------------------------------
__global__ __launch_bounds__(256) void transpose_cast_kernel(
    const float* __restrict__ in, unsigned short* __restrict__ out, int K, int N) {
  __shared__ float tile[32][33];
  const int bn = blockIdx.x * 32;   // over N
  const int bk = blockIdx.y * 32;   // over K
  const int tx = threadIdx.x, ty = threadIdx.y;  // 32 x 8
#pragma unroll
  for (int i = 0; i < 32; i += 8)
    tile[ty + i][tx] = in[(size_t)(bk + ty + i) * N + bn + tx];
  __syncthreads();
#pragma unroll
  for (int i = 0; i < 32; i += 8)
    out[(size_t)(bn + ty + i) * K + bk + tx] = f2bf(tile[tx][ty + i]);
}

// ---------------------------------------------------------------------------
// Generic bf16 GEMM: C[M][N] = A[M][K] * Bt[N][K]^T, f32 accumulate.
// Block tile 128x128x32, 8 waves (4 along M x 2 along N), each wave 32x64.
// Double-buffered LDS, register-staged global loads so vmem latency overlaps
// the WMMA burst of the previous tile.
// ---------------------------------------------------------------------------
template <bool OUT_BF16>
__global__ __launch_bounds__(256) void gemm_bf16_kernel(
    const unsigned short* __restrict__ A,
    const unsigned short* __restrict__ Bt,
    void* __restrict__ Cout, int M, int N, int K) {
  __shared__ __align__(16) unsigned short sA[2][128 * 32];
  __shared__ __align__(16) unsigned short sB[2][128 * 32];

  const int bm = blockIdx.y * 128;
  const int bn = blockIdx.x * 128;
  const int tid  = threadIdx.x;
  const int wave = tid >> 5;
  const int lane = tid & 31;
  const int g  = lane >> 4;
  const int ln = lane & 15;
  const int wm = (wave & 3) * 32;   // 4 waves along M
  const int wn = (wave >> 2) * 64;  // 2 waves along N

  const v8f vzero = {0.f, 0.f, 0.f, 0.f, 0.f, 0.f, 0.f, 0.f};
  v8f acc[2][4];
#pragma unroll
  for (int i = 0; i < 2; ++i)
#pragma unroll
    for (int j = 0; j < 4; ++j) acc[i][j] = vzero;

  const int lr = tid >> 1;          // 0..127 : tile row this thread copies
  const int lc = (tid & 1) * 16;    // 0 or 16
  const unsigned short* ga = A  + (size_t)(bm + lr) * K + lc;
  const unsigned short* gb = Bt + (size_t)(bn + lr) * K + lc;
  const int soff = lr * 32 + lc;

  // prologue: tile 0 -> LDS buf0, tile 1 -> registers
  u32x4 ra0 = *(const u32x4*)(const void*)(ga);
  u32x4 ra1 = *(const u32x4*)(const void*)(ga + 8);
  u32x4 rb0 = *(const u32x4*)(const void*)(gb);
  u32x4 rb1 = *(const u32x4*)(const void*)(gb + 8);
  *(u32x4*)(void*)&sA[0][soff]     = ra0;
  *(u32x4*)(void*)&sA[0][soff + 8] = ra1;
  *(u32x4*)(void*)&sB[0][soff]     = rb0;
  *(u32x4*)(void*)&sB[0][soff + 8] = rb1;
  if (32 < K) {
    ra0 = *(const u32x4*)(const void*)(ga + 32);
    ra1 = *(const u32x4*)(const void*)(ga + 40);
    rb0 = *(const u32x4*)(const void*)(gb + 32);
    rb1 = *(const u32x4*)(const void*)(gb + 40);
  }
  __syncthreads();

  int cur = 0;
  for (int kb = 0; kb < K; kb += 32) {
    // batch all fragment loads, then 8 back-to-back WMMAs
    Frag a0, a1, b0, b1, b2, b3;
    load_frag(a0, sA[cur], 32, wm, 0, lane);
    load_frag(a1, sA[cur], 32, wm + 16, 0, lane);
    load_frag(b0, sB[cur], 32, wn, 0, lane);
    load_frag(b1, sB[cur], 32, wn + 16, 0, lane);
    load_frag(b2, sB[cur], 32, wn + 32, 0, lane);
    load_frag(b3, sB[cur], 32, wn + 48, 0, lane);
    acc[0][0] = wmma_bf16(a0, b0, acc[0][0]);
    acc[0][1] = wmma_bf16(a0, b1, acc[0][1]);
    acc[0][2] = wmma_bf16(a0, b2, acc[0][2]);
    acc[0][3] = wmma_bf16(a0, b3, acc[0][3]);
    acc[1][0] = wmma_bf16(a1, b0, acc[1][0]);
    acc[1][1] = wmma_bf16(a1, b1, acc[1][1]);
    acc[1][2] = wmma_bf16(a1, b2, acc[1][2]);
    acc[1][3] = wmma_bf16(a1, b3, acc[1][3]);
    __syncthreads();
    if (kb + 32 < K) {
      // stage tile kb+32 (already in regs) into the other buffer
      *(u32x4*)(void*)&sA[cur ^ 1][soff]     = ra0;
      *(u32x4*)(void*)&sA[cur ^ 1][soff + 8] = ra1;
      *(u32x4*)(void*)&sB[cur ^ 1][soff]     = rb0;
      *(u32x4*)(void*)&sB[cur ^ 1][soff + 8] = rb1;
      if (kb + 64 < K) {  // fetch tile kb+64; wait lands next iteration
        ra0 = *(const u32x4*)(const void*)(ga + kb + 64);
        ra1 = *(const u32x4*)(const void*)(ga + kb + 72);
        rb0 = *(const u32x4*)(const void*)(gb + kb + 64);
        rb1 = *(const u32x4*)(const void*)(gb + kb + 72);
      }
      __syncthreads();
      cur ^= 1;
    }
  }

#pragma unroll
  for (int i = 0; i < 2; ++i)
#pragma unroll
    for (int j = 0; j < 4; ++j)
#pragma unroll
      for (int r = 0; r < 8; ++r) {
        const int m = bm + wm + 16 * i + 8 * g + r;  // C layout: vgpr r, lane half g
        const int n = bn + wn + 16 * j + ln;
        const float val = acc[i][j][r];
        if constexpr (OUT_BF16)
          ((unsigned short*)Cout)[(size_t)m * N + n] = f2bf(val);
        else
          ((float*)Cout)[(size_t)m * N + n] = val;
      }
}

// ---------------------------------------------------------------------------
// RoPE + pack: q/k -> head-major [B*H][T][192] (rope on last 64 dims),
//              v   -> transposed  [B*H][192][T]
// ---------------------------------------------------------------------------
__global__ __launch_bounds__(192) void rope_pack_kernel(
    const unsigned short* __restrict__ qc,  // [M][2048]
    const unsigned short* __restrict__ qr,  // [M][1024]
    const unsigned short* __restrict__ kc,  // [M][2048]
    const unsigned short* __restrict__ kr,  // [M][1024]
    const unsigned short* __restrict__ v,   // [M][3072]
    unsigned short* __restrict__ qhm, unsigned short* __restrict__ khm,
    unsigned short* __restrict__ vt, int T) {
  const int bt = blockIdx.x;
  const int b = bt / T, t = bt % T;
  const int d = threadIdx.x;  // 0..191

  float c = 0.f, s = 0.f;
  int pairbase = 0;
  if (d >= 128) {
    const int i = d - 128;           // 0..63
    const int pair = i >> 1;         // 0..31
    const float inv = __expf(-((float)(2 * pair) / 64.f) * 9.2103403719761836f);
    const float ang = (float)t * inv;
    c = __cosf(ang);
    s = __sinf(ang);
    pairbase = 2 * pair;
  }

  for (int h = 0; h < 16; ++h) {
    const int bh = b * 16 + h;
    const size_t o = ((size_t)bh * T + t) * 192 + d;
    unsigned short qv, kv;
    if (d < 128) {
      qv = qc[(size_t)bt * 2048 + h * 128 + d];
      kv = kc[(size_t)bt * 2048 + h * 128 + d];
    } else {
      const int i = d - 128;
      const size_t base = (size_t)bt * 1024 + h * 64 + pairbase;
      const float x1q = bf2f(qr[base]), x2q = bf2f(qr[base + 1]);
      const float x1k = bf2f(kr[base]), x2k = bf2f(kr[base + 1]);
      const float oq = (i & 1) ? (x1q * s + x2q * c) : (x1q * c - x2q * s);
      const float ok = (i & 1) ? (x1k * s + x2k * c) : (x1k * c - x2k * s);
      qv = f2bf(oq);
      kv = f2bf(ok);
    }
    qhm[o] = qv;
    khm[o] = kv;
    vt[((size_t)bh * 192 + d) * T + t] = v[(size_t)bt * 3072 + h * 192 + d];
  }
}

// ---------------------------------------------------------------------------
// Flash attention per (b,h): 64-query blocks, 4 waves x 16 rows, causal,
// online softmax.  S = Q K^T and O += P V both via v_wmma_f32_16x16x32_bf16.
// ---------------------------------------------------------------------------
__global__ __launch_bounds__(128) void mla_attn_kernel(
    const unsigned short* __restrict__ Q,   // [B*H][T][192]
    const unsigned short* __restrict__ Kh,  // [B*H][T][192]
    const unsigned short* __restrict__ Vt,  // [B*H][192][T]
    unsigned short* __restrict__ O,         // [B][T][H*192]
    int T) {
  __shared__ __align__(16) unsigned short sQ[64 * 192];
  __shared__ __align__(16) unsigned short sK[64 * 192];
  __shared__ __align__(16) unsigned short sV[192 * 64];
  __shared__ __align__(16) unsigned short sP[4][16 * 64];  // per-wave scratch

  const int bh = blockIdx.y;
  const int b = bh >> 4, h = bh & 15;
  const int qb = blockIdx.x;  // 64-row query block
  const int tid = threadIdx.x;
  const int wave = tid >> 5, lane = tid & 31;
  const int g = lane >> 4, ln = lane & 15;

  // Q block is 64 contiguous rows of 192
  const unsigned short* Qg = Q + ((size_t)bh * T + qb * 64) * 192;
  for (int e = tid * 8; e < 64 * 192; e += 128 * 8)
    *(u32x4*)(void*)&sQ[e] = *(const u32x4*)(const void*)&Qg[e];

  const v8f vzero = {0.f, 0.f, 0.f, 0.f, 0.f, 0.f, 0.f, 0.f};
  v8f acc[12];
#pragma unroll
  for (int d = 0; d < 12; ++d) acc[d] = vzero;
  float mrow[8], lrow[8];
#pragma unroll
  for (int r = 0; r < 8; ++r) { mrow[r] = -3.0e38f; lrow[r] = 0.0f; }

  const float scale = 0.07216878364870323f;  // 1/sqrt(192)

  for (int j = 0; j <= qb; ++j) {
    // stage K block (contiguous) and V^T block (strided rows) into LDS
    const unsigned short* Kg = Kh + ((size_t)bh * T + j * 64) * 192;
    for (int e = tid * 8; e < 64 * 192; e += 128 * 8)
      *(u32x4*)(void*)&sK[e] = *(const u32x4*)(const void*)&Kg[e];
    const unsigned short* Vg = Vt + (size_t)bh * 192 * T + j * 64;
    for (int u = tid; u < 192 * 8; u += 128) {
      const int row = u >> 3, cc = (u & 7) * 8;
      *(u32x4*)(void*)&sV[row * 64 + cc] =
          *(const u32x4*)(const void*)&Vg[(size_t)row * T + cc];
    }
    if (j < qb) __builtin_prefetch(Kg + 64 * 192, 0, 0);  // next K block
    __syncthreads();

    // ---- S = Q K^T : 16x64 per wave, 24 WMMAs (batched frag loads)
    v8f s[4];
#pragma unroll
    for (int nt = 0; nt < 4; ++nt) s[nt] = vzero;
#pragma unroll
    for (int kd = 0; kd < 6; ++kd) {
      Frag af, b0, b1, b2, b3;
      load_frag(af, sQ, 192, wave * 16, kd * 32, lane);
      load_frag(b0, sK, 192, 0,  kd * 32, lane);
      load_frag(b1, sK, 192, 16, kd * 32, lane);
      load_frag(b2, sK, 192, 32, kd * 32, lane);
      load_frag(b3, sK, 192, 48, kd * 32, lane);
      s[0] = wmma_bf16(af, b0, s[0]);
      s[1] = wmma_bf16(af, b1, s[1]);
      s[2] = wmma_bf16(af, b2, s[2]);
      s[3] = wmma_bf16(af, b3, s[3]);
    }

    // ---- scale + causal mask (diagonal block only)
    const int qrow0 = qb * 64 + wave * 16 + 8 * g;  // + r
    const int kcol0 = j * 64 + ln;                  // + nt*16
#pragma unroll
    for (int nt = 0; nt < 4; ++nt)
#pragma unroll
      for (int r = 0; r < 8; ++r) {
        float val = s[nt][r] * scale;
        if (j == qb && (kcol0 + nt * 16) > (qrow0 + r)) val = -3.0e38f;
        s[nt][r] = val;
      }

    // ---- online softmax (row reductions within 16-lane halves)
    float corr[8], mnew[8];
#pragma unroll
    for (int r = 0; r < 8; ++r) {
      float mx = fmaxf(fmaxf(s[0][r], s[1][r]), fmaxf(s[2][r], s[3][r]));
#pragma unroll
      for (int off = 8; off > 0; off >>= 1) mx = fmaxf(mx, __shfl_xor(mx, off, 32));
      mnew[r] = fmaxf(mrow[r], mx);
      corr[r] = __expf(mrow[r] - mnew[r]);
      mrow[r] = mnew[r];
    }
    unsigned short* sPw = sP[wave];
#pragma unroll
    for (int r = 0; r < 8; ++r) {
      float rs = 0.0f;
#pragma unroll
      for (int nt = 0; nt < 4; ++nt) {
        const float p = __expf(s[nt][r] - mnew[r]);
        rs += p;
        sPw[(8 * g + r) * 64 + nt * 16 + ln] = f2bf(p);  // C-layout -> row-major
      }
#pragma unroll
      for (int off = 8; off > 0; off >>= 1) rs += __shfl_xor(rs, off, 32);
      lrow[r] = lrow[r] * corr[r] + rs;
    }
#pragma unroll
    for (int d = 0; d < 12; ++d)
#pragma unroll
      for (int r = 0; r < 8; ++r) acc[d][r] *= corr[r];

    // ---- O += P V : A = P (16x64), B = V^T tile (N=d major), 24 WMMAs
#pragma unroll
    for (int ks = 0; ks < 2; ++ks) {
      Frag pf;
      load_frag(pf, sPw, 64, 0, ks * 32, lane);
#pragma unroll
      for (int ndg = 0; ndg < 3; ++ndg) {
        Frag v0, v1, v2, v3;
        load_frag(v0, sV, 64, (ndg * 4 + 0) * 16, ks * 32, lane);
        load_frag(v1, sV, 64, (ndg * 4 + 1) * 16, ks * 32, lane);
        load_frag(v2, sV, 64, (ndg * 4 + 2) * 16, ks * 32, lane);
        load_frag(v3, sV, 64, (ndg * 4 + 3) * 16, ks * 32, lane);
        acc[ndg * 4 + 0] = wmma_bf16(pf, v0, acc[ndg * 4 + 0]);
        acc[ndg * 4 + 1] = wmma_bf16(pf, v1, acc[ndg * 4 + 1]);
        acc[ndg * 4 + 2] = wmma_bf16(pf, v2, acc[ndg * 4 + 2]);
        acc[ndg * 4 + 3] = wmma_bf16(pf, v3, acc[ndg * 4 + 3]);
      }
    }
    __syncthreads();
  }

  // epilogue: normalize and scatter to [B][T][H*192]
  const int trow = qb * 64 + wave * 16 + 8 * g;
#pragma unroll
  for (int nd = 0; nd < 12; ++nd)
#pragma unroll
    for (int r = 0; r < 8; ++r) {
      const float val = acc[nd][r] / lrow[r];
      O[((size_t)b * T + trow + r) * 3072 + h * 192 + nd * 16 + ln] = f2bf(val);
    }
}

// ---------------------------------------------------------------------------
// host launcher
// ---------------------------------------------------------------------------
extern "C" void kernel_launch(void* const* d_in, const int* in_sizes, int n_in,
                              void* d_out, int out_size, void* d_ws, size_t ws_size,
                              hipStream_t stream) {
  (void)in_sizes; (void)n_in; (void)out_size; (void)ws_size;
  const int B = 4, T = 1024, C = 2048, H = 16, DN = 128, DR = 64, D = 192;
  const int QR = 1536, KVR = 512;
  const int M = B * T;  // 4096 token rows

  const float* x   = (const float*)d_in[0];
  const float* wqd = (const float*)d_in[1];
  const float* wqn = (const float*)d_in[2];
  const float* wqr = (const float*)d_in[3];
  const float* wkd = (const float*)d_in[4];
  const float* wvu = (const float*)d_in[5];
  const float* wkn = (const float*)d_in[6];
  const float* wkr = (const float*)d_in[7];
  const float* wo  = (const float*)d_in[8];
  float* out = (float*)d_out;

  unsigned short* ws = (unsigned short*)d_ws;
  size_t off = 0;
  auto alloc = [&](size_t elems) {
    unsigned short* p = ws + off;
    off += (elems + 127) & ~(size_t)127;   // 256B-aligned chunks
    return p;
  };

  unsigned short* xb    = alloc((size_t)M * C);
  unsigned short* wqd_t = alloc((size_t)QR * C);
  unsigned short* wqn_t = alloc((size_t)(H * DN) * QR);
  unsigned short* wqr_t = alloc((size_t)(H * DR) * QR);
  unsigned short* wkd_t = alloc((size_t)KVR * C);
  unsigned short* wvu_t = alloc((size_t)(H * D) * KVR);
  unsigned short* wkn_t = alloc((size_t)(H * DN) * KVR);
  unsigned short* wkr_t = alloc((size_t)(H * DR) * KVR);
  unsigned short* wo_t  = alloc((size_t)C * (H * D));
  unsigned short* qlat  = alloc((size_t)M * QR);
  unsigned short* kvlat = alloc((size_t)M * KVR);
  unsigned short* qc    = alloc((size_t)M * H * DN);
  unsigned short* qr_   = alloc((size_t)M * H * DR);
  unsigned short* kc    = alloc((size_t)M * H * DN);
  unsigned short* kr_   = alloc((size_t)M * H * DR);
  unsigned short* vbuf  = alloc((size_t)M * H * D);
  unsigned short* qhm   = alloc((size_t)B * H * T * D);
  unsigned short* khm   = alloc((size_t)B * H * T * D);
  unsigned short* vt    = alloc((size_t)B * H * D * T);
  unsigned short* ao    = alloc((size_t)M * H * D);

  const dim3 tb(32, 8);
  // casts / transposes to bf16
  cast_f32_bf16_kernel<<<(M * C) / 1024, 256, 0, stream>>>(x, xb, M * C);
  transpose_cast_kernel<<<dim3(QR / 32, C / 32), tb, 0, stream>>>(wqd, wqd_t, C, QR);
  transpose_cast_kernel<<<dim3((H * DN) / 32, QR / 32), tb, 0, stream>>>(wqn, wqn_t, QR, H * DN);
  transpose_cast_kernel<<<dim3((H * DR) / 32, QR / 32), tb, 0, stream>>>(wqr, wqr_t, QR, H * DR);
  transpose_cast_kernel<<<dim3(KVR / 32, C / 32), tb, 0, stream>>>(wkd, wkd_t, C, KVR);
  transpose_cast_kernel<<<dim3((H * D) / 32, KVR / 32), tb, 0, stream>>>(wvu, wvu_t, KVR, H * D);
  transpose_cast_kernel<<<dim3((H * DN) / 32, KVR / 32), tb, 0, stream>>>(wkn, wkn_t, KVR, H * DN);
  transpose_cast_kernel<<<dim3((H * DR) / 32, KVR / 32), tb, 0, stream>>>(wkr, wkr_t, KVR, H * DR);
  transpose_cast_kernel<<<dim3(C / 32, (H * D) / 32), tb, 0, stream>>>(wo, wo_t, H * D, C);

  // stage 1: latents
  gemm_bf16_kernel<true><<<dim3(QR / 128, M / 128), 256, 0, stream>>>(xb, wqd_t, qlat, M, QR, C);
  gemm_bf16_kernel<true><<<dim3(KVR / 128, M / 128), 256, 0, stream>>>(xb, wkd_t, kvlat, M, KVR, C);
  // stage 2: up-projections
  gemm_bf16_kernel<true><<<dim3((H * DN) / 128, M / 128), 256, 0, stream>>>(qlat, wqn_t, qc, M, H * DN, QR);
  gemm_bf16_kernel<true><<<dim3((H * DR) / 128, M / 128), 256, 0, stream>>>(qlat, wqr_t, qr_, M, H * DR, QR);
  gemm_bf16_kernel<true><<<dim3((H * D) / 128, M / 128), 256, 0, stream>>>(kvlat, wvu_t, vbuf, M, H * D, KVR);
  gemm_bf16_kernel<true><<<dim3((H * DN) / 128, M / 128), 256, 0, stream>>>(kvlat, wkn_t, kc, M, H * DN, KVR);
  gemm_bf16_kernel<true><<<dim3((H * DR) / 128, M / 128), 256, 0, stream>>>(kvlat, wkr_t, kr_, M, H * DR, KVR);
  // rope + head-major pack + V transpose
  rope_pack_kernel<<<M, 192, 0, stream>>>(qc, qr_, kc, kr_, vbuf, qhm, khm, vt, T);
  // flash attention per (b,h)
  mla_attn_kernel<<<dim3(T / 64, B * H), 128, 0, stream>>>(qhm, khm, vt, ao, T);
  // output projection -> f32
  gemm_bf16_kernel<false><<<dim3(C / 128, M / 128), 256, 0, stream>>>(ao, wo_t, out, M, C, H * D);
}
